// H_BLSTM_76416058130988
// MI455X (gfx1250) — matrix operand order repaired and checked by reading, not compiled, not run on hardware
//
#include <hip/hip_runtime.h>
#include <hip/hip_bf16.h>

// ---------------------------------------------------------------------------
// Types for CDNA5 WMMA (wave32, 16x16x32 bf16 -> f32)
// ---------------------------------------------------------------------------
typedef unsigned short u16;
typedef __bf16 bf16_t;
typedef bf16_t v16bf __attribute__((ext_vector_type(16)));
typedef float  v8f   __attribute__((ext_vector_type(8)));
typedef u16    v16u  __attribute__((ext_vector_type(16)));
typedef u16    v8u   __attribute__((ext_vector_type(8)));

#define BSZ 256   // batch
#define TSZ 256   // time steps
#define INSZ 128
#define HSZ 512
#define G5 2560   // 5*H

__device__ inline u16 f2bf(float f) {
    unsigned u = __builtin_bit_cast(unsigned, f);
    u += 0x7FFFu + ((u >> 16) & 1u);          // round to nearest even
    return (u16)(u >> 16);
}

__device__ inline v16bf mk_frag(v8u lo, v8u hi) {
    v16u c;
#pragma unroll
    for (int i = 0; i < 8; ++i) { c[i] = lo[i]; c[i + 8] = hi[i]; }
    return __builtin_bit_cast(v16bf, c);
}

// A-matrix fragment (16x32 bf16, row-major source). rowp = &row[0] for this
// lane's row (row = lane%16). lane<16 holds K = kb+0..15,
// lane>=16 holds K = kb+16..31 (matches packed B).
__device__ inline v16bf load_Afrag(const u16* rowp, int lane, int kbase) {
    const int ko = (lane < 16) ? 0 : 16;
    const u16* p = rowp + kbase + ko;
    v8u lo = *(const v8u*)p;
    v8u hi = *(const v8u*)(p + 8);
    return mk_frag(lo, hi);
}

// B-matrix fragment from pre-packed layout: per (ntile,ktile) block of
// 32 lanes x 16 contiguous bf16 (lane<16: K=0..15, lane>=16: K=16..31; N=lane%16).
__device__ inline v16bf load_Bfrag(const u16* blk, int lane) {
    const u16* p = blk + lane * 16;
    v8u lo = *(const v8u*)p;
    v8u hi = *(const v8u*)(p + 8);
    return mk_frag(lo, hi);
}

__device__ inline v8f wmma_bf16(v16bf a, v16bf b, v8f c) {
    return __builtin_amdgcn_wmma_f32_16x16x32_bf16(false, a, false, b,
                                                   (short)0, c, false, false);
}

__device__ inline float sigm(float x) { return 1.0f / (1.0f + __expf(-x)); }

// ---------------------------------------------------------------------------
// Pack kernels
// ---------------------------------------------------------------------------
__global__ void pack_x_kernel(const float* __restrict__ in, u16* __restrict__ out, int n) {
    int i = blockIdx.x * 256 + threadIdx.x;
    if (i < n) out[i] = f2bf(in[i]);
}

// Pack f32 [K,N] row-major weight into WMMA-B fragment blocks:
// out[((ntile*nkt + ktile)*32 + lane)*16 + e] = W[(kt*32 + (lane<16?0:16) + e)*N + ntile*16 + lane%16]
__global__ void pack_B_kernel(const float* __restrict__ W, u16* __restrict__ out,
                              int K, int N) {
    int tid = blockIdx.x * 256 + threadIdx.x;
    int nkt = K >> 5;
    int total = (N >> 4) * nkt * 32;
    if (tid >= total) return;
    int lane = tid & 31;
    int kt = (tid >> 5) % nkt;
    int nt = tid / (32 * nkt);
    int n = nt * 16 + (lane & 15);
    int kb = kt * 32 + ((lane < 16) ? 0 : 16);
    u16* o = out + (size_t)tid * 16;
#pragma unroll
    for (int e = 0; e < 16; ++e)
        o[e] = f2bf(W[(size_t)(kb + e) * N + n]);
}

// ---------------------------------------------------------------------------
// Input projection GEMM, layer 1:  xg[t,b,:] = x[b,t,:] @ Wx + bias
// A = x_bf16 [B,T,128]; output [T,B,2560] f32. Block: 8 waves, one 16x16
// output tile per wave per ktile-loop; grid (4096, 20).
// ---------------------------------------------------------------------------
__global__ __launch_bounds__(256)
void gemm_xg1_kernel(const u16* __restrict__ Abf, const u16* __restrict__ Bp,
                     const float* __restrict__ bias, float* __restrict__ out) {
    const int lane = threadIdx.x & 31, wave = threadIdx.x >> 5;
    const int nlo = lane & 15, rhalf = (lane >> 4) * 8;
    const int ntile = blockIdx.y * 8 + wave;
    const int col = ntile * 16 + nlo;
    const int r0 = blockIdx.x * 16;            // output rows r = t*B + b
    const int t = r0 / BSZ, b0 = r0 % BSZ;
    const u16* arow = Abf + ((size_t)(b0 + nlo) * TSZ + t) * INSZ;

    v8f acc;
    const float bv = bias[col];
#pragma unroll
    for (int e = 0; e < 8; ++e) acc[e] = bv;

#pragma unroll
    for (int kt = 0; kt < 4; ++kt) {
        v16bf a = load_Afrag(arow, lane, kt * 32);
        v16bf b = load_Bfrag(Bp + ((size_t)(ntile * 4 + kt)) * 512, lane);
        acc = wmma_bf16(a, b, acc);
    }
#pragma unroll
    for (int e = 0; e < 8; ++e)
        out[(size_t)(r0 + rhalf + e) * G5 + col] = acc[e];
}

// ---------------------------------------------------------------------------
// Input projection GEMM, layer 2: A = [hs_f | hs_b] bf16 (each [T,B,512],
// row index = t*B+b matches output row). K = 1024.
// ---------------------------------------------------------------------------
__global__ __launch_bounds__(256)
void gemm_xg2_kernel(const u16* __restrict__ hsf, const u16* __restrict__ hsb,
                     const u16* __restrict__ Bp, const float* __restrict__ bias,
                     float* __restrict__ out) {
    const int lane = threadIdx.x & 31, wave = threadIdx.x >> 5;
    const int nlo = lane & 15, rhalf = (lane >> 4) * 8;
    const int ntile = blockIdx.y * 8 + wave;
    const int col = ntile * 16 + nlo;
    const int r0 = blockIdx.x * 16;
    const u16* rowf = hsf + (size_t)(r0 + nlo) * HSZ;
    const u16* rowb = hsb + (size_t)(r0 + nlo) * HSZ;

    v8f acc;
    const float bv = bias[col];
#pragma unroll
    for (int e = 0; e < 8; ++e) acc[e] = bv;

#pragma unroll 4
    for (int kt = 0; kt < 32; ++kt) {
        v16bf a = (kt < 16) ? load_Afrag(rowf, lane, kt * 32)
                            : load_Afrag(rowb, lane, (kt - 16) * 32);
        v16bf b = load_Bfrag(Bp + ((size_t)(ntile * 32 + kt)) * 512, lane);
        acc = wmma_bf16(a, b, acc);
    }
#pragma unroll
    for (int e = 0; e < 8; ++e)
        out[(size_t)(r0 + rhalf + e) * G5 + col] = acc[e];
}

// ---------------------------------------------------------------------------
// Recurrent scan: grid = 32 WGs (dir = blk>>4, 16-row batch tile = blk&15),
// 512 threads = 16 waves. Each wave owns 2 j-tiles x 5 gates (10 WMMA acc).
// States k/kp live in registers; h broadcast via LDS bf16 tile each step.
// ---------------------------------------------------------------------------
__global__ __launch_bounds__(512)
void scan_kernel(const float* __restrict__ xg_f, const float* __restrict__ xg_b,
                 const u16* __restrict__ WhP_f, const u16* __restrict__ WhP_b,
                 u16* __restrict__ hs_f, u16* __restrict__ hs_b,
                 float* __restrict__ hT, int store_hs) {
    const int dir = blockIdx.x >> 4;
    const int m0 = (blockIdx.x & 15) * 16;
    const float* xg = dir ? xg_b : xg_f;
    const u16* WhP = dir ? WhP_b : WhP_f;
    u16* hs = dir ? hs_b : hs_f;

    const int lane = threadIdx.x & 31;
    const int wave = threadIdx.x >> 5;
    const int nlo = lane & 15;
    const int rhalf = (lane >> 4) * 8;
    const int jt0 = wave * 2;

    __shared__ u16 hsh[16][520];               // bf16 h tile, padded rows
    for (int i = threadIdx.x; i < 16 * 520; i += 512)
        ((u16*)hsh)[i] = 0;

    v8f k_st[2], kp_st[2], h_cur[2];
#pragma unroll
    for (int j = 0; j < 2; ++j) {
#pragma unroll
        for (int e = 0; e < 8; ++e) { k_st[j][e] = 0.f; kp_st[j][e] = 0.f; h_cur[j][e] = 0.f; }
    }

    for (int step = 0; step < TSZ; ++step) {
        const int t = dir ? (TSZ - 1 - step) : step;
        __syncthreads();                       // h tile ready for all waves

        v8f acc[5][2];
#pragma unroll
        for (int g = 0; g < 5; ++g) {
#pragma unroll
            for (int j = 0; j < 2; ++j) {
                const int col = g * HSZ + (jt0 + j) * 16 + nlo;
#pragma unroll
                for (int e = 0; e < 8; ++e) {
                    const int row = m0 + rhalf + e;
                    acc[g][j][e] = xg[(size_t)(t * BSZ + row) * G5 + col];
                }
            }
        }

#pragma unroll 2
        for (int kt = 0; kt < 16; ++kt) {
            v16bf a = load_Afrag(&hsh[nlo][0], lane, kt * 32);
#pragma unroll
            for (int g = 0; g < 5; ++g) {
#pragma unroll
                for (int j = 0; j < 2; ++j) {
                    const int ntile = g * 32 + jt0 + j;
                    v16bf b = load_Bfrag(WhP + ((size_t)(ntile * 16 + kt)) * 512, lane);
                    acc[g][j] = wmma_bf16(a, b, acc[g][j]);
                }
            }
        }
        __syncthreads();                       // all reads of h done

        // gate nonlinearity + state update (pure per-lane register math)
#pragma unroll
        for (int j = 0; j < 2; ++j) {
            const int jh = (jt0 + j) * 16 + nlo;
#pragma unroll
            for (int e = 0; e < 8; ++e) {
                const float f = sigm(acc[0][j][e]);
                const float i = sigm(acc[1][j][e]);
                const float o = sigm(acc[2][j][e]);
                const float u = sigm(acc[3][j][e]);
                const float kp = u * tanhf(acc[4][j][e]) + (1.f - u) * kp_st[j][e];
                const float kk = f * k_st[j][e] + i * kp;
                const float h = o * tanhf(kk);
                kp_st[j][e] = kp; k_st[j][e] = kk; h_cur[j][e] = h;
                const int row = rhalf + e;
                const u16 hb = f2bf(h);
                hsh[row][jh] = hb;
                if (store_hs)
                    hs[(size_t)(t * BSZ + m0 + row) * HSZ + jh] = hb;
            }
        }
    }

    if (!store_hs) {                           // layer 2: emit final hidden state
#pragma unroll
        for (int j = 0; j < 2; ++j) {
            const int jh = (jt0 + j) * 16 + nlo;
#pragma unroll
            for (int e = 0; e < 8; ++e) {
                const int row = m0 + rhalf + e;
                hT[(size_t)row * 1024 + dir * HSZ + jh] = h_cur[j][e];
            }
        }
    }
}

// ---------------------------------------------------------------------------
// Head: relu(hT @ W1 + b1) @ W2 + b2
// ---------------------------------------------------------------------------
__global__ __launch_bounds__(256)
void head1_kernel(const float* __restrict__ hT, const float* __restrict__ W1,
                  const float* __restrict__ b1, float* __restrict__ hidden) {
    int idx = blockIdx.x * 256 + threadIdx.x;  // 256*512
    int b = idx >> 9, j = idx & 511;
    float s = b1[j];
    for (int c = 0; c < 1024; ++c)
        s += hT[b * 1024 + c] * W1[c * 512 + j];
    hidden[idx] = fmaxf(s, 0.f);
}

__global__ __launch_bounds__(256)
void head2_kernel(const float* __restrict__ hidden, const float* __restrict__ W2,
                  const float* __restrict__ b2, float* __restrict__ out) {
    int b = blockIdx.x, tid = threadIdx.x;
    float s = hidden[b * 512 + tid] * W2[tid] +
              hidden[b * 512 + 256 + tid] * W2[256 + tid];
    __shared__ float red[256];
    red[tid] = s;
    __syncthreads();
    for (int off = 128; off; off >>= 1) {
        if (tid < off) red[tid] += red[tid + off];
        __syncthreads();
    }
    if (tid == 0) out[b] = red[0] + b2[0];
}

// ---------------------------------------------------------------------------
// Launch
// ---------------------------------------------------------------------------
extern "C" void kernel_launch(void* const* d_in, const int* in_sizes, int n_in,
                              void* d_out, int out_size, void* d_ws, size_t ws_size,
                              hipStream_t stream) {
    const float* x      = (const float*)d_in[0];
    const float* f1_Wx  = (const float*)d_in[1];
    const float* f1_Wh  = (const float*)d_in[2];
    const float* f1_b   = (const float*)d_in[3];
    const float* b1_Wx  = (const float*)d_in[4];
    const float* b1_Wh  = (const float*)d_in[5];
    const float* b1_b   = (const float*)d_in[6];
    const float* f2_Wx  = (const float*)d_in[7];
    const float* f2_Wh  = (const float*)d_in[8];
    const float* f2_b   = (const float*)d_in[9];
    const float* b2_Wx  = (const float*)d_in[10];
    const float* b2_Wh  = (const float*)d_in[11];
    const float* b2_b   = (const float*)d_in[12];
    const float* headW1 = (const float*)d_in[13];
    const float* headb1 = (const float*)d_in[14];
    const float* headW2 = (const float*)d_in[15];
    const float* headb2 = (const float*)d_in[16];
    float* out = (float*)d_out;

    char* w = (char*)d_ws;
    size_t off = 0;
    auto carve = [&](size_t bytes) {
        void* p = w + off;
        off = (off + bytes + 255) & ~(size_t)255;
        return p;
    };
    const size_t nx = (size_t)BSZ * TSZ * INSZ;
    u16* x_bf  = (u16*)carve(nx * 2);
    u16* WxP1f = (u16*)carve((size_t)160 * 4 * 512 * 2);
    u16* WxP1b = (u16*)carve((size_t)160 * 4 * 512 * 2);
    u16* WhP1f = (u16*)carve((size_t)160 * 16 * 512 * 2);
    u16* WhP1b = (u16*)carve((size_t)160 * 16 * 512 * 2);
    u16* WhP2f = (u16*)carve((size_t)160 * 16 * 512 * 2);
    u16* WhP2b = (u16*)carve((size_t)160 * 16 * 512 * 2);
    u16* WxP2f = (u16*)carve((size_t)160 * 32 * 512 * 2);
    u16* WxP2b = (u16*)carve((size_t)160 * 32 * 512 * 2);
    u16* hs_f  = (u16*)carve((size_t)TSZ * BSZ * HSZ * 2);
    u16* hs_b  = (u16*)carve((size_t)TSZ * BSZ * HSZ * 2);
    float* hT     = (float*)carve((size_t)BSZ * 1024 * 4);
    float* hidden = (float*)carve((size_t)BSZ * HSZ * 4);
    float* xg_f = (float*)carve((size_t)TSZ * BSZ * G5 * 4);  // reused layer1/layer2
    float* xg_b = (float*)carve((size_t)TSZ * BSZ * G5 * 4);

    // ---- pack inputs/weights to bf16 WMMA layouts ----
    pack_x_kernel<<<(nx + 255) / 256, 256, 0, stream>>>(x, x_bf, (int)nx);
    pack_B_kernel<<<80,  256, 0, stream>>>(f1_Wx, WxP1f, 128, G5);
    pack_B_kernel<<<80,  256, 0, stream>>>(b1_Wx, WxP1b, 128, G5);
    pack_B_kernel<<<320, 256, 0, stream>>>(f1_Wh, WhP1f, 512, G5);
    pack_B_kernel<<<320, 256, 0, stream>>>(b1_Wh, WhP1b, 512, G5);
    pack_B_kernel<<<640, 256, 0, stream>>>(f2_Wx, WxP2f, 1024, G5);
    pack_B_kernel<<<640, 256, 0, stream>>>(b2_Wx, WxP2b, 1024, G5);
    pack_B_kernel<<<320, 256, 0, stream>>>(f2_Wh, WhP2f, 512, G5);
    pack_B_kernel<<<320, 256, 0, stream>>>(b2_Wh, WhP2b, 512, G5);

    // ---- layer 1: input projections + bidirectional scans ----
    dim3 ggrid(4096, 20);
    gemm_xg1_kernel<<<ggrid, 256, 0, stream>>>(x_bf, WxP1f, f1_b, xg_f);
    gemm_xg1_kernel<<<ggrid, 256, 0, stream>>>(x_bf, WxP1b, b1_b, xg_b);
    scan_kernel<<<32, 512, 0, stream>>>(xg_f, xg_b, WhP1f, WhP1b, hs_f, hs_b, hT, 1);

    // ---- layer 2: input projections (A = [hs_f|hs_b]) + scans ----
    gemm_xg2_kernel<<<ggrid, 256, 0, stream>>>(hs_f, hs_b, WxP2f, f2_b, xg_f);
    gemm_xg2_kernel<<<ggrid, 256, 0, stream>>>(hs_f, hs_b, WxP2b, b2_b, xg_b);
    scan_kernel<<<32, 512, 0, stream>>>(xg_f, xg_b, WhP2f, WhP2b, hs_f, hs_b, hT, 0);

    // ---- head ----
    head1_kernel<<<512, 256, 0, stream>>>(hT, headW1, headb1, hidden);
    head2_kernel<<<256, 256, 0, stream>>>(hidden, headW2, headb2, out);
}
